// MultiHeadAttention_12421045420258
// MI455X (gfx1250) — matrix operand order repaired and checked by reading, not compiled, not run on hardware
//
#include <hip/hip_runtime.h>
#include <hip/hip_bf16.h>

// ---------------------------------------------------------------------------
// MultiHeadAttention for MI455X (gfx1250): bf16 WMMA everywhere, f32 accum.
// S=4096, D_MODEL=1024, H=16, d_k=64.
// ---------------------------------------------------------------------------

#define SLEN 4096
#define DMODEL 1024
#define NHEAD 16
#define DK 64
#define MWORDS (SLEN / 32)   // bitmask words per query row

typedef __attribute__((ext_vector_type(16))) __bf16 v16bf;
typedef __attribute__((ext_vector_type(8)))  float  v8f;

union Frag16 {
    v16bf v;
    uint4 q[2];
};

__device__ __forceinline__ v8f wmma_bf16(v16bf a, v16bf b, v8f c) {
    // D = A(16x32 bf16) * B(32x16 bf16) + C(16x16 f32)
    return __builtin_amdgcn_wmma_f32_16x16x32_bf16(
        /*neg_a=*/false, a, /*neg_b=*/false, b,
        /*c_mod=*/(short)0, c, /*reuse_a=*/false, /*reuse_b=*/false);
}

// ---------------------------------------------------------------------------
// fp32 -> bf16 conversion (n multiple of 4)
// ---------------------------------------------------------------------------
__global__ void f32_to_bf16_kernel(const float* __restrict__ in,
                                   __bf16* __restrict__ out, int n) {
    int i = (blockIdx.x * blockDim.x + threadIdx.x) * 4;
    if (i < n) {
        float4 v = *(const float4*)(in + i);
        out[i + 0] = (__bf16)v.x;
        out[i + 1] = (__bf16)v.y;
        out[i + 2] = (__bf16)v.z;
        out[i + 3] = (__bf16)v.w;
    }
}

// ---------------------------------------------------------------------------
// int32 mask [S][S] -> 1 bit per element [S][S/32]  (64MB -> 2MB, L2-resident)
// ---------------------------------------------------------------------------
__global__ void pack_mask_kernel(const int* __restrict__ mask,
                                 unsigned int* __restrict__ bits, int nwords) {
    int w = blockIdx.x * blockDim.x + threadIdx.x;
    if (w < nwords) {
        const int* p = mask + (size_t)w * 32;
        unsigned int b = 0;
#pragma unroll
        for (int j = 0; j < 32; j += 4) {
            int4 m = *(const int4*)(p + j);
            b |= (m.x != 0 ? 1u : 0u) << (j + 0);
            b |= (m.y != 0 ? 1u : 0u) << (j + 1);
            b |= (m.z != 0 ? 1u : 0u) << (j + 2);
            b |= (m.w != 0 ? 1u : 0u) << (j + 3);
        }
        bits[w] = b;
    }
}

// ---------------------------------------------------------------------------
// C[M,N] = A[M,K](bf16) @ W[N,K]^T(bf16) + bias
// Block: 256 threads = 8 waves arranged 4(M) x 2(N); block tile 128x128;
// each wave computes a 32x64 tile (2x4 WMMA fragments, A reused 4x, B 2x).
// mode 0: f32 row-major [M,N]
// mode 1: bf16 head-split [H][S][64]   (col n -> head n>>6, d n&63)
// mode 2: bf16 transposed [H][64][S]   (for V)
// ---------------------------------------------------------------------------
__global__ __launch_bounds__(256) void gemm_bf16_kernel(
    const __bf16* __restrict__ A, const __bf16* __restrict__ Bw,
    const float* __restrict__ bias, void* __restrict__ out,
    int M, int N, int K, int mode)
{
    const int lane = threadIdx.x & 31;
    const int wave = threadIdx.x >> 5;
    const int wm = wave >> 1, wn = wave & 1;
    const int half = lane >> 4, l15 = lane & 15;
    const int m0 = blockIdx.y * 128 + wm * 32;
    const int n0 = blockIdx.x * 128 + wn * 64;

    v8f zero = {};
    v8f acc[2][4];
#pragma unroll
    for (int fm = 0; fm < 2; ++fm)
#pragma unroll
        for (int fn = 0; fn < 4; ++fn) acc[fm][fn] = zero;

    for (int kk = 0; kk < K; kk += 32) {
        Frag16 a[2], b[4];
#pragma unroll
        for (int f = 0; f < 2; ++f) {
            // A fragment: lane = row m, half selects K-subrange {0..7,16..23}/{8..15,24..31}
            const __bf16* ap = A + (size_t)(m0 + f * 16 + l15) * K + kk;
            a[f].q[0] = *(const uint4*)(ap + half * 8);
            a[f].q[1] = *(const uint4*)(ap + 16 + half * 8);
        }
#pragma unroll
        for (int f = 0; f < 4; ++f) {
            // B fragment: lane = col n; half selects K-subrange {0..7,16..23}/{8..15,24..31}.
            // B[k][n] = W[n][k] -> contiguous along W's row.
            const __bf16* bp = Bw + (size_t)(n0 + f * 16 + l15) * K + kk;
            b[f].q[0] = *(const uint4*)(bp + half * 8);
            b[f].q[1] = *(const uint4*)(bp + 16 + half * 8);
        }
#pragma unroll
        for (int fm = 0; fm < 2; ++fm)
#pragma unroll
            for (int fn = 0; fn < 4; ++fn)
                acc[fm][fn] = wmma_bf16(a[fm].v, b[fn].v, acc[fm][fn]);
    }

#pragma unroll
    for (int fm = 0; fm < 2; ++fm)
#pragma unroll
        for (int fn = 0; fn < 4; ++fn) {
            int n = n0 + fn * 16 + l15;
            float bv = bias[n];
#pragma unroll
            for (int i = 0; i < 8; ++i) {
                int m = m0 + fm * 16 + i + 8 * half;   // C-frag row mapping
                float val = acc[fm][fn][i] + bv;
                if (mode == 0) {
                    ((float*)out)[(size_t)m * N + n] = val;
                } else if (mode == 1) {
                    ((__bf16*)out)[(size_t)(n >> 6) * M * DK + (size_t)m * DK + (n & 63)] =
                        (__bf16)val;
                } else {
                    ((__bf16*)out)[(size_t)(n >> 6) * DK * M + (size_t)(n & 63) * M + m] =
                        (__bf16)val;
                }
            }
        }
}

// ---------------------------------------------------------------------------
// Flash attention per head.
// Grid: (S/64, H). Block: 128 threads = 4 waves; wave w owns 16 query rows.
// Streams 64-key chunks (two 32-key WMMA subtiles per LDS stage):
// K/V staged in LDS (4x cross-wave reuse), S = Q K^T via WMMA, online softmax
// with bitmask, P relayout via per-wave LDS, P V via WMMA. Next chunk is
// prefetched (global_prefetch_b8) while the current one computes.
// ---------------------------------------------------------------------------
__global__ __launch_bounds__(128) void attn_kernel(
    const __bf16* __restrict__ Qh,          // [H][S][64]
    const __bf16* __restrict__ Kh,          // [H][S][64]
    const __bf16* __restrict__ Vt,          // [H][64][S]
    const unsigned int* __restrict__ mbits, // [S][S/32]
    __bf16* __restrict__ attn_out)          // [S][1024] bf16 (heads concat)
{
    __shared__ __align__(16) __bf16 Ks[64 * 64];     // [key][d]     8KB
    __shared__ __align__(16) __bf16 Vs[64 * 64];     // [d][key]     8KB
    __shared__ __align__(16) __bf16 Ps[4][16 * 32];  // per-wave P   4KB

    const int tid = threadIdx.x;
    const int lane = tid & 31, wave = tid >> 5;
    const int half = lane >> 4, l15 = lane & 15;
    const int h = blockIdx.y;
    const int q0 = blockIdx.x * 64 + wave * 16;

    // Q fragments for this wave's 16 rows: invariant across the key loop.
    Frag16 aq[2];
    {
        const __bf16* qp = Qh + ((size_t)h * SLEN + q0 + l15) * DK;
#pragma unroll
        for (int ks = 0; ks < 2; ++ks) {
            aq[ks].q[0] = *(const uint4*)(qp + ks * 32 + half * 8);
            aq[ks].q[1] = *(const uint4*)(qp + ks * 32 + 16 + half * 8);
        }
    }

    v8f zero = {};
    float mrow[8], lrow[8];
    v8f o[4];
#pragma unroll
    for (int i = 0; i < 8; ++i) { mrow[i] = -1e30f; lrow[i] = 0.0f; }
#pragma unroll
    for (int dt = 0; dt < 4; ++dt) o[dt] = zero;

    // Cooperative-load geometry: thread covers 32 contiguous elements (64B).
    const int rk = tid >> 1;            // row in 64-row tile
    const int ck = (tid & 1) * 32;      // column offset

    for (int kc = 0; kc < SLEN; kc += 64) {
        // --- stage K[64][64] and V^T[64][64] tiles into LDS ---
        {
            const uint4* sk = (const uint4*)(Kh + ((size_t)h * SLEN + kc + rk) * DK + ck);
            uint4* dk = (uint4*)(Ks + rk * 64 + ck);
            dk[0] = sk[0]; dk[1] = sk[1]; dk[2] = sk[2]; dk[3] = sk[3];

            const uint4* sv = (const uint4*)(Vt + ((size_t)h * DK + rk) * SLEN + kc + ck);
            uint4* dv = (uint4*)(Vs + rk * 64 + ck);
            dv[0] = sv[0]; dv[1] = sv[1]; dv[2] = sv[2]; dv[3] = sv[3];
        }
        // prefetch next chunk while this one computes
        if (kc + 64 < SLEN) {
            __builtin_prefetch(Kh + ((size_t)h * SLEN + kc + 64 + rk) * DK + ck, 0, 1);
            __builtin_prefetch(Vt + ((size_t)h * DK + rk) * SLEN + kc + 64 + ck, 0, 1);
        }
        __syncthreads();

#pragma unroll
        for (int sub = 0; sub < 2; ++sub) {
            const int kb = kc + sub * 32;

            // S = Q K^T : two 16x16 C-frags over 32 keys, 2 K-steps each.
            v8f s[2];
            s[0] = zero; s[1] = zero;
#pragma unroll
            for (int nt = 0; nt < 2; ++nt)
#pragma unroll
                for (int ks = 0; ks < 2; ++ks) {
                    Frag16 b;  // B[k=d][n=key] = K[key][d] -> contiguous in Ks row
                    const __bf16* kp =
                        Ks + (sub * 32 + nt * 16 + l15) * 64 + ks * 32 + half * 8;
                    b.q[0] = *(const uint4*)kp; b.q[1] = *(const uint4*)(kp + 16);
                    s[nt] = wmma_bf16(aq[ks].v, b.v, s[nt]);
                }

            // scale + mask via bitmask (one b32 per row covers all 32 keys)
#pragma unroll
            for (int i = 0; i < 8; ++i) {
                int qrow = q0 + i + 8 * half;
                unsigned int mw = mbits[(size_t)qrow * MWORDS + (kb >> 5)];
                float x0 = s[0][i] * 0.125f;   // 1/sqrt(64)
                float x1 = s[1][i] * 0.125f;
                if (!((mw >> l15) & 1u))        x0 = -1e9f;
                if (!((mw >> (16 + l15)) & 1u)) x1 = -1e9f;
                s[0][i] = x0;
                s[1][i] = x1;
            }

            // Online softmax: row reductions live within each 16-lane half.
            float c[8];
#pragma unroll
            for (int i = 0; i < 8; ++i) {
                float v = fmaxf(s[0][i], s[1][i]);
#pragma unroll
                for (int off = 1; off < 16; off <<= 1)
                    v = fmaxf(v, __shfl_xor(v, off, 32));
                float mnew = fmaxf(mrow[i], v);
                c[i] = __expf(mrow[i] - mnew);
                float p0 = __expf(s[0][i] - mnew);
                float p1 = __expf(s[1][i] - mnew);
                s[0][i] = p0; s[1][i] = p1;
                float rs = p0 + p1;
#pragma unroll
                for (int off = 1; off < 16; off <<= 1)
                    rs += __shfl_xor(rs, off, 32);
                lrow[i] = lrow[i] * c[i] + rs;
                mrow[i] = mnew;
            }
#pragma unroll
            for (int dt = 0; dt < 4; ++dt)
#pragma unroll
                for (int i = 0; i < 8; ++i) o[dt][i] *= c[i];

            // P: C-frag layout -> A-frag layout through per-wave LDS (in-order DS).
#pragma unroll
            for (int nt = 0; nt < 2; ++nt)
#pragma unroll
                for (int i = 0; i < 8; ++i)
                    Ps[wave][(i + 8 * half) * 32 + nt * 16 + l15] = (__bf16)s[nt][i];

            Frag16 pa;
            {
                const __bf16* pp = &Ps[wave][l15 * 32];
                pa.q[0] = *(const uint4*)(pp + half * 8);
                pa.q[1] = *(const uint4*)(pp + 16 + half * 8);
            }

            // O += P V : B[k=key][n=d] = Vt[d][key] -> contiguous in Vs row
#pragma unroll
            for (int dt = 0; dt < 4; ++dt) {
                Frag16 b;
                const __bf16* vp =
                    Vs + (dt * 16 + l15) * 64 + sub * 32 + half * 8;
                b.q[0] = *(const uint4*)vp; b.q[1] = *(const uint4*)(vp + 16);
                o[dt] = wmma_bf16(pa.v, b.v, o[dt]);
            }
        }
        __syncthreads();
    }

    // Normalize and emit bf16 [S][1024] with heads concatenated.
#pragma unroll
    for (int dt = 0; dt < 4; ++dt)
#pragma unroll
        for (int i = 0; i < 8; ++i) {
            int qrow = q0 + i + 8 * half;
            int d = dt * 16 + l15;
            attn_out[(size_t)qrow * DMODEL + h * DK + d] = (__bf16)(o[dt][i] / lrow[i]);
        }
}

// ---------------------------------------------------------------------------
// Host launcher
// ---------------------------------------------------------------------------
extern "C" void kernel_launch(void* const* d_in, const int* in_sizes, int n_in,
                              void* d_out, int out_size, void* d_ws, size_t ws_size,
                              hipStream_t stream) {
    const float* q    = (const float*)d_in[0];
    const float* k    = (const float*)d_in[1];
    const float* v    = (const float*)d_in[2];
    const int*   mask = (const int*)d_in[3];
    const float* Wq = (const float*)d_in[4];
    const float* bq = (const float*)d_in[5];
    const float* Wk = (const float*)d_in[6];
    const float* bk = (const float*)d_in[7];
    const float* Wv = (const float*)d_in[8];
    const float* bv = (const float*)d_in[9];
    const float* Wo = (const float*)d_in[10];
    const float* bo = (const float*)d_in[11];

    const size_t ACT = (size_t)SLEN * DMODEL;   // 4M elems
    const size_t WEL = (size_t)DMODEL * DMODEL; // 1M elems
    const int    NWORDS = SLEN * MWORDS;        // 512K bitmask words

    __bf16* ws  = (__bf16*)d_ws;
    __bf16* xq  = ws;            // bf16 inputs
    __bf16* xk  = xq + ACT;
    __bf16* xv  = xk + ACT;
    __bf16* wqb = xv + ACT;      // bf16 weights
    __bf16* wkb = wqb + WEL;
    __bf16* wvb = wkb + WEL;
    __bf16* wob = wvb + WEL;
    __bf16* Qh  = wob + WEL;     // [H][S][64]
    __bf16* Kh  = Qh + ACT;      // [H][S][64]
    __bf16* Vt  = Kh + ACT;      // [H][64][S]
    unsigned int* mbits = (unsigned int*)(Vt + ACT);  // [S][S/32]
    __bf16* attnb = xq;          // reuse: xq dead after Q projection

    // --- fp32 -> bf16 conversions + mask packing ---
    f32_to_bf16_kernel<<<ACT / 1024, 256, 0, stream>>>(q, xq, (int)ACT);
    f32_to_bf16_kernel<<<ACT / 1024, 256, 0, stream>>>(k, xk, (int)ACT);
    f32_to_bf16_kernel<<<ACT / 1024, 256, 0, stream>>>(v, xv, (int)ACT);
    f32_to_bf16_kernel<<<WEL / 1024, 256, 0, stream>>>(Wq, wqb, (int)WEL);
    f32_to_bf16_kernel<<<WEL / 1024, 256, 0, stream>>>(Wk, wkb, (int)WEL);
    f32_to_bf16_kernel<<<WEL / 1024, 256, 0, stream>>>(Wv, wvb, (int)WEL);
    f32_to_bf16_kernel<<<WEL / 1024, 256, 0, stream>>>(Wo, wob, (int)WEL);
    pack_mask_kernel<<<NWORDS / 256, 256, 0, stream>>>(mask, mbits, NWORDS);

    // --- projections (M=4096, N=1024, K=1024), 128x128 block tiles ---
    dim3 gg(DMODEL / 128, SLEN / 128);
    gemm_bf16_kernel<<<gg, 256, 0, stream>>>(xq, wqb, bq, Qh, SLEN, DMODEL, DMODEL, 1);
    gemm_bf16_kernel<<<gg, 256, 0, stream>>>(xk, wkb, bk, Kh, SLEN, DMODEL, DMODEL, 1);
    gemm_bf16_kernel<<<gg, 256, 0, stream>>>(xv, wvb, bv, Vt, SLEN, DMODEL, DMODEL, 2);

    // --- attention ---
    dim3 ga(SLEN / 64, NHEAD);
    attn_kernel<<<ga, 128, 0, stream>>>(Qh, Kh, Vt, mbits, attnb);

    // --- output projection -> fp32 d_out ---
    gemm_bf16_kernel<<<gg, 256, 0, stream>>>(attnb, wob, bo, d_out, SLEN, DMODEL, DMODEL, 0);
}
